// MultiHeadAttention_58102317580570
// MI455X (gfx1250) — matrix operand order repaired and checked by reading, not compiled, not run on hardware
//
#include <hip/hip_runtime.h>
#include <math.h>

#define DEVINL __device__ __forceinline__

typedef _Float16 h16_t;
typedef __attribute__((ext_vector_type(16))) _Float16 v16h;
typedef __attribute__((ext_vector_type(8)))  _Float16 v8h;
typedef __attribute__((ext_vector_type(8)))  float    v8f;
typedef int i4v __attribute__((vector_size(16)));

constexpr int Bn = 2, Cn = 2048, Mn = 1024, Hn = 16, Kd = 64, Vd = 64;

union V16U { v16h v; uint4 u[2]; };
union V8U  { v8h  v; uint4 u;    };

// ---------------- CDNA5 async global->LDS copy (ASYNCcnt path) ----------------
DEVINL void async_cp16(h16_t* lds, const h16_t* g) {
#if __has_builtin(__builtin_amdgcn_global_load_async_to_lds_b128)
  __builtin_amdgcn_global_load_async_to_lds_b128(
      (__attribute__((address_space(1))) i4v*)(void*)(h16_t*)g,
      (__attribute__((address_space(3))) i4v*)(void*)lds, 0, 0);
#else
  unsigned loff = (unsigned)(unsigned long)(__attribute__((address_space(3))) h16_t*)lds;
  asm volatile("global_load_async_to_lds_b128 %0, %1, off"
               :: "v"(loff), "v"(g) : "memory");
#endif
}

DEVINL void wait_async0() {
#if __has_builtin(__builtin_amdgcn_s_wait_asynccnt)
  __builtin_amdgcn_s_wait_asynccnt(0);
#else
  asm volatile("s_wait_asynccnt 0x0" ::: "memory");
#endif
}

// Stage a 64-row x 32-col f16 panel (4KB) into LDS; 256 threads, 16B each.
DEVINL void stage_64x32(h16_t* lds, const h16_t* gbase, long rowStride, int t) {
  int r  = t >> 2;
  int ch = (t & 3) * 8;
  async_cp16(lds + r * 32 + ch, gbase + (long)r * rowStride + ch);
}
// Stage a 32-row x 64-col f16 panel (4KB) into LDS; 256 threads, 16B each.
DEVINL void stage_32x64(h16_t* lds, const h16_t* gbase, long rowStride, int t) {
  int r  = t >> 3;
  int ch = (t & 7) * 8;
  async_cp16(lds + r * 64 + ch, gbase + (long)r * rowStride + ch);
}

// A-operand (16x32 f16, MxK): lane = row (mod 16); vector elems 0..15 hold
// K = 2*e8+0..15 (contiguous), where e8 = 8*(lane>>4).
DEVINL v16h load_a32(const h16_t* row, int e8) {
  V16U r;
  r.u[0] = *(const uint4*)(row + 2 * e8);
  r.u[1] = *(const uint4*)(row + 2 * e8 + 8);
  return r.v;
}
// B-operand (32x16 f16, KxN): lane = column (mod 16); elems hold 16 contiguous
// K starting at e16 = 16*(lane>>4). Column-major source => one 32B load.
DEVINL v16h load_b32(const h16_t* col) {
  V16U r;
  r.u[0] = *(const uint4*)(col);
  r.u[1] = *(const uint4*)(col + 8);
  return r.v;
}
DEVINL v8f wmma16(v16h a, v16h b, v8f c) {
  return __builtin_amdgcn_wmma_f32_16x16x32_f16(
      false, a, false, b, (short)0, c, false, false);
}

// ---------------- conversion kernels ----------------
__global__ void cvt_copy(const float* __restrict__ src, h16_t* __restrict__ dst, int n) {
  int i = blockIdx.x * blockDim.x + threadIdx.x;
  int stride = gridDim.x * blockDim.x;
  for (; i < n; i += stride) dst[i] = (h16_t)src[i];
}

// src[b][r][n] (f32) -> dst[b][n][r] (f16)
__global__ void cvt_transpose(const float* __restrict__ src, h16_t* __restrict__ dst,
                              int batch, int R, int N) {
  long total = (long)batch * R * N;
  long i = (long)blockIdx.x * blockDim.x + threadIdx.x;
  long stride = (long)gridDim.x * blockDim.x;
  for (; i < total; i += stride) {
    long bb = i / ((long)R * N);
    long j  = i % ((long)R * N);
    long n  = j / R;
    long r  = j % R;
    dst[i] = (h16_t)src[bb * R * N + r * N + n];
  }
}

// ---------------- QKV projection ----------------
// grid: ((p*B + b)*H + h)*(C/128) + ctile ; 8 waves, wave = 16 rows x 64 cols.
// Weight panel (64 out-cols x 32 m) staged through LDS, double buffered.
__global__ __launch_bounds__(256) void proj_kernel(
    const h16_t* __restrict__ q_h, const h16_t* __restrict__ kv_h,
    const h16_t* __restrict__ wqT, const h16_t* __restrict__ wkT,
    const h16_t* __restrict__ wvT,
    h16_t* __restrict__ Qp, h16_t* __restrict__ Kp, h16_t* __restrict__ VpT) {
  __shared__ alignas(16) h16_t wbuf[2][64 * 32];

  int bid = blockIdx.x;
  int ct = bid % (Cn / 128); bid /= (Cn / 128);
  int h  = bid % Hn;          bid /= Hn;
  int b  = bid % Bn;          bid /= Bn;
  int p  = bid;               // 0=Q, 1=K, 2=V

  const h16_t* X  = (p == 0) ? q_h : kv_h;
  const h16_t* WT = (p == 0) ? wqT : ((p == 1) ? wkT : wvT);

  int tid  = threadIdx.x;
  int wave = tid >> 5;
  int lane = tid & 31;
  int lmod = lane & 15;
  int e8   = (lane >> 4) * 8;
  int e16  = (lane >> 4) * 16;

  int c = ct * 128 + wave * 16 + lmod;
  const h16_t* arow  = X + ((long)(b * Cn + c)) * Mn;
  const h16_t* wbase = WT + (long)h * 64 * Mn; // WT[h][n][m]

  stage_64x32(wbuf[0], wbase, Mn, tid);
  wait_async0();
  __syncthreads();

  v8f acc[4] = {};
  for (int m0 = 0; m0 < Mn; m0 += 32) {
    int pb = (m0 >> 5) & 1;
    if (m0 + 32 < Mn) stage_64x32(wbuf[1 - pb], wbase + m0 + 32, Mn, tid);

    v16h a = load_a32(arow + m0, e8);
#pragma unroll
    for (int t = 0; t < 4; ++t) {
      const h16_t* brow = wbuf[pb] + (t * 16 + lmod) * 32 + e16;
      acc[t] = wmma16(a, load_b32(brow), acc[t]);
    }
    wait_async0();
    __syncthreads();
  }

  int bh = b * Hn + h;
  if (p < 2) {
    h16_t* O = (p == 0) ? Qp : Kp;
#pragma unroll
    for (int t = 0; t < 4; ++t)
#pragma unroll
      for (int j = 0; j < 8; ++j) {
        int cr = ct * 128 + wave * 16 + j + e8;
        O[((long)(bh * Cn + cr)) * 64 + t * 16 + lmod] = (h16_t)acc[t][j];
      }
  } else {
    // Store V transposed: VpT[b][h][v][c]
#pragma unroll
    for (int t = 0; t < 4; ++t) {
      V8U s;
#pragma unroll
      for (int j = 0; j < 8; ++j) s.v[j] = (h16_t)acc[t][j];
      int v = t * 16 + lmod;
      long addr = ((long)(bh * 64 + v)) * Cn + ct * 128 + wave * 16 + e8;
      *(uint4*)(VpT + addr) = s.u;
    }
  }
}

// ---------------- flash attention ----------------
// grid: (b*H + h)*(C/128) + dtile ; wave owns 16 queries, streams 2048 keys in
// blocks of 32. K (32x64) and V^T (64x32) blocks staged via async->LDS, double
// buffered; S computed as [c rows][d lanes]; softmax per lane (=query).
__global__ __launch_bounds__(256) void attn_kernel(
    const h16_t* __restrict__ Qp, const h16_t* __restrict__ Kp,
    const h16_t* __restrict__ VpT, h16_t* __restrict__ PRE) {
  __shared__ alignas(16) h16_t kbuf[2][32 * 64];
  __shared__ alignas(16) h16_t vbuf[2][64 * 32];

  int bid = blockIdx.x;
  int dt = bid % (Cn / 128); bid /= (Cn / 128);
  int h  = bid % Hn;          bid /= Hn;
  int b  = bid;
  int bh = b * Hn + h;

  int tid  = threadIdx.x;
  int wave = tid >> 5;
  int lane = tid & 31;
  int lmod = lane & 15;
  int e8   = (lane >> 4) * 8;
  int e16  = (lane >> 4) * 16;
  bool low = (lane < 16);

  int d0 = dt * 128 + wave * 16;
  const float scale = 0.125f; // 1/sqrt(64)

  const h16_t* kgbase = Kp + (long)(bh * Cn) * 64;   // rows = keys, stride 64
  const h16_t* vgbase = VpT + (long)(bh * 64) * Cn;  // rows = v,    stride Cn

  // Q as B-operand (registers, reused for all key blocks)
  const h16_t* qrow = Qp + ((long)(bh * Cn + d0 + lmod)) * 64;
  v16h bq0 = load_b32(qrow + e16);       // k = 0..31
  v16h bq1 = load_b32(qrow + 32 + e16);  // k = 32..63

  stage_32x64(kbuf[0], kgbase, 64, tid);
  stage_64x32(vbuf[0], vgbase, Cn, tid);
  wait_async0();
  __syncthreads();

  v8f pre[4] = {};
  float mrow = -1e30f, ssum = 0.f;

  constexpr int NBLK = Cn / 32;
  for (int i = 0; i < NBLK; ++i) {
    int pb = i & 1;
    if (i + 1 < NBLK) {
      int c1 = (i + 1) * 32;
      stage_32x64(kbuf[1 - pb], kgbase + (long)c1 * 64, 64, tid);
      stage_64x32(vbuf[1 - pb], vgbase + c1, Cn, tid);
    }

    const h16_t* krow0 = kbuf[pb] + lmod * 64;
    const h16_t* krow1 = kbuf[pb] + (16 + lmod) * 64;
    v8f S0 = {}, S1 = {};
    S0 = wmma16(load_a32(krow0,      e8), bq0, S0);
    S0 = wmma16(load_a32(krow0 + 32, e8), bq1, S0);
    S1 = wmma16(load_a32(krow1,      e8), bq0, S1);
    S1 = wmma16(load_a32(krow1 + 32, e8), bq1, S1);

    float bmax = -1e30f;
#pragma unroll
    for (int j = 0; j < 8; ++j) {
      S0[j] *= scale; S1[j] *= scale;
      bmax = fmaxf(bmax, fmaxf(S0[j], S1[j]));
    }
    bmax = fmaxf(bmax, __shfl_xor(bmax, 16, 32)); // merge halves (same query)
    float nmax = fmaxf(mrow, bmax);
    float corr = __expf(mrow - nmax);
    float P0[8], P1[8];
    float psum = 0.f;
#pragma unroll
    for (int j = 0; j < 8; ++j) {
      P0[j] = __expf(S0[j] - nmax);
      P1[j] = __expf(S1[j] - nmax);
      psum += P0[j] + P1[j];
    }
    psum += __shfl_xor(psum, 16, 32);
    ssum = ssum * corr + psum;
    mrow = nmax;
#pragma unroll
    for (int t = 0; t < 4; ++t)
#pragma unroll
      for (int j = 0; j < 8; ++j) pre[t][j] *= corr;

    // Reshape P (two D-layout tiles) into one B-operand (32 keys x 16 queries).
    V16U bp;
#pragma unroll
    for (int j = 0; j < 8; ++j) {
      float o0 = __shfl_xor(P0[j], 16, 32);
      float o1 = __shfl_xor(P1[j], 16, 32);
      bp.v[j]     = (h16_t)(low ? P0[j] : o1);
      bp.v[j + 8] = (h16_t)(low ? o0 : P1[j]);
    }

    // pre^T[v][d] += V^T[v][c] * P[c][d]
#pragma unroll
    for (int vt = 0; vt < 4; ++vt) {
      const h16_t* vrow = vbuf[pb] + (vt * 16 + lmod) * 32;
      pre[vt] = wmma16(load_a32(vrow, e8), bp.v, pre[vt]);
    }

    wait_async0();
    __syncthreads();
  }

  float inv = 1.0f / ssum;
  // PRE[b][c=query][h*64+v] f16, contiguous in v per lane.
  long base = ((long)(b * Cn + d0 + lmod)) * (Hn * Vd) + h * 64;
#pragma unroll
  for (int vt = 0; vt < 4; ++vt) {
    V8U s;
#pragma unroll
    for (int j = 0; j < 8; ++j) s.v[j] = (h16_t)(pre[vt][j] * inv);
    *(uint4*)(PRE + base + vt * 16 + e8) = s.u;
  }
}

// ---------------- output projection ----------------
// out[b][c][m] = sum_{hv} PRE[b][c][hv] * woT[m][hv]; woT panel staged via LDS.
__global__ __launch_bounds__(256) void outproj_kernel(
    const h16_t* __restrict__ PRE, const h16_t* __restrict__ woT,
    float* __restrict__ out) {
  __shared__ alignas(16) h16_t wbuf[2][64 * 32];

  int bid = blockIdx.x;
  int mt = bid % (Mn / 64);  bid /= (Mn / 64);
  int ct = bid % (Cn / 128); bid /= (Cn / 128);
  int b  = bid;

  int tid  = threadIdx.x;
  int wave = tid >> 5;
  int lane = tid & 31;
  int lmod = lane & 15;
  int e8   = (lane >> 4) * 8;
  int e16  = (lane >> 4) * 16;

  int c = ct * 128 + wave * 16 + lmod;
  const h16_t* arow  = PRE + ((long)(b * Cn + c)) * (Hn * Vd);
  const h16_t* wbase = woT + (long)(mt * 64) * (Hn * Vd); // rows = out cols m

  stage_64x32(wbuf[0], wbase, Hn * Vd, tid);
  wait_async0();
  __syncthreads();

  v8f acc[4] = {};
  for (int k0 = 0; k0 < Hn * Vd; k0 += 32) {
    int pb = (k0 >> 5) & 1;
    if (k0 + 32 < Hn * Vd) stage_64x32(wbuf[1 - pb], wbase + k0 + 32, Hn * Vd, tid);

    v16h a = load_a32(arow + k0, e8);
#pragma unroll
    for (int t = 0; t < 4; ++t) {
      const h16_t* brow = wbuf[pb] + (t * 16 + lmod) * 32 + e16;
      acc[t] = wmma16(a, load_b32(brow), acc[t]);
    }
    wait_async0();
    __syncthreads();
  }
#pragma unroll
  for (int t = 0; t < 4; ++t)
#pragma unroll
    for (int j = 0; j < 8; ++j) {
      int cr = ct * 128 + wave * 16 + j + e8;
      out[((long)(b * Cn + cr)) * Mn + mt * 64 + t * 16 + lmod] = acc[t][j];
    }
}

// ---------------- host launcher ----------------
extern "C" void kernel_launch(void* const* d_in, const int* in_sizes, int n_in,
                              void* d_out, int out_size, void* d_ws, size_t ws_size,
                              hipStream_t stream) {
  const float* kvinput = (const float*)d_in[0];
  const float* qinput  = (const float*)d_in[1];
  const float* wq      = (const float*)d_in[2];
  const float* wk      = (const float*)d_in[3];
  const float* wv      = (const float*)d_in[4];
  const float* wo      = (const float*)d_in[5];
  float* out = (float*)d_out;

  char* w = (char*)d_ws;
  size_t off = 0;
  auto take = [&](size_t bytes) -> void* {
    void* p = w + off;
    off += (bytes + 255) & ~(size_t)255;
    return p;
  };
  h16_t* q_h  = (h16_t*)take((size_t)Bn * Cn * Mn * 2);
  h16_t* kv_h = (h16_t*)take((size_t)Bn * Cn * Mn * 2);
  h16_t* wqT  = (h16_t*)take((size_t)Hn * Kd * Mn * 2);
  h16_t* wkT  = (h16_t*)take((size_t)Hn * Kd * Mn * 2);
  h16_t* wvT  = (h16_t*)take((size_t)Hn * Vd * Mn * 2);
  h16_t* woT  = (h16_t*)take((size_t)Mn * Hn * Vd * 2);
  h16_t* Qp   = (h16_t*)take((size_t)Bn * Hn * Cn * Kd * 2);
  h16_t* Kp   = (h16_t*)take((size_t)Bn * Hn * Cn * Kd * 2);
  h16_t* VpT  = (h16_t*)take((size_t)Bn * Hn * Vd * Cn * 2);
  h16_t* PRE  = (h16_t*)take((size_t)Bn * Cn * Hn * Vd * 2);

  cvt_copy<<<1024, 256, 0, stream>>>(qinput, q_h, Bn * Cn * Mn);
  cvt_copy<<<1024, 256, 0, stream>>>(kvinput, kv_h, Bn * Cn * Mn);
  cvt_transpose<<<512, 256, 0, stream>>>(wq, wqT, Hn, Mn, Kd);
  cvt_transpose<<<512, 256, 0, stream>>>(wk, wkT, Hn, Mn, Kd);
  cvt_transpose<<<512, 256, 0, stream>>>(wv, wvT, Hn, Mn, Vd);
  cvt_transpose<<<512, 256, 0, stream>>>(wo, woT, 1, Hn * Vd, Mn);

  proj_kernel<<<3 * Bn * Hn * (Cn / 128), 256, 0, stream>>>(
      q_h, kv_h, wqT, wkT, wvT, Qp, Kp, VpT);
  attn_kernel<<<Bn * Hn * (Cn / 128), 256, 0, stream>>>(Qp, Kp, VpT, PRE);
  outproj_kernel<<<Bn * (Cn / 128) * (Mn / 64), 256, 0, stream>>>(PRE, woT, out);
}
